// SelfAttention_9826885173977
// MI455X (gfx1250) — matrix operand order repaired and checked by reading, not compiled, not run on hardware
//
#include <hip/hip_runtime.h>
#include <math.h>

#define Bsz 4
#define Ssz 2048
#define Esz 1024

typedef __attribute__((ext_vector_type(16))) __bf16 v16bf;
typedef __attribute__((ext_vector_type(8)))  float  v8f;
typedef __attribute__((ext_vector_type(4)))  int    v4i_;

// Block tile 128x128, k-step 32, 256 threads (8 waves, wave32).
constexpr int BLK   = 128;
constexpr int BLK_K = 32;
constexpr int KP    = 40;   // LDS row pitch in bf16 (32 data + 8 pad) = 80B, 16B aligned
constexpr int KPU   = 20;   // pitch in u32
constexpr int KPU4  = 5;    // pitch in uint4
constexpr int TILE4 = BLK * KPU4;   // uint4 elems per tile buffer

// Async global->LDS DMA path (gfx1250): guard so compilation never regresses.
#if defined(__AMDGCN__) && __has_builtin(__builtin_amdgcn_global_load_async_to_lds_b128)
#define HAVE_ASYNC 1
#else
#define HAVE_ASYNC 0
#endif

// ---------- helpers ----------
__device__ __forceinline__ unsigned short f2bf(float f) {
  unsigned int u = __float_as_uint(f);
  unsigned int r = 0x7FFFu + ((u >> 16) & 1u);   // round-to-nearest-even
  return (unsigned short)((u + r) >> 16);
}
__device__ __forceinline__ unsigned int pk2(float a, float b) {
#if defined(__AMDGCN__) && __has_builtin(__builtin_amdgcn_cvt_pk_bf16_f32)
  typedef __attribute__((ext_vector_type(2))) __bf16 v2bf;
  v2bf r = __builtin_amdgcn_cvt_pk_bf16_f32(a, b);
  return __builtin_bit_cast(unsigned int, r);
#else
  return (unsigned int)f2bf(a) | ((unsigned int)f2bf(b) << 16);
#endif
}

__device__ __forceinline__ void async_wait0() {
#if defined(__AMDGCN__) && __has_builtin(__builtin_amdgcn_s_wait_asynccnt)
  __builtin_amdgcn_s_wait_asynccnt(0);
#elif defined(__AMDGCN__)
  asm volatile("s_wait_asynccnt 0" ::: "memory");
#endif
}

// Stage a 128x32 tile from f32 global (converting to bf16) into padded LDS.
// 256 threads: 2 threads/row, 16 elems each.
__device__ __forceinline__ void stage_f32(uint4* lds4, const float* __restrict__ src,
                                          int pitch, int tid) {
  int row  = tid >> 1;
  int half = tid & 1;
  const float4* sp = (const float4*)(src + (size_t)row * (size_t)pitch + half * 16);
  float4 x0 = sp[0], x1 = sp[1], x2 = sp[2], x3 = sp[3];
  uint4 o0, o1;
  o0.x = pk2(x0.x, x0.y); o0.y = pk2(x0.z, x0.w);
  o0.z = pk2(x1.x, x1.y); o0.w = pk2(x1.z, x1.w);
  o1.x = pk2(x2.x, x2.y); o1.y = pk2(x2.z, x2.w);
  o1.z = pk2(x3.x, x3.y); o1.w = pk2(x3.z, x3.w);
  lds4[row * KPU4 + half * 2 + 0] = o0;
  lds4[row * KPU4 + half * 2 + 1] = o1;
}

// Synchronous bf16 tile staging (fallback path).
__device__ __forceinline__ void stage_bf16(uint4* lds4, const unsigned short* __restrict__ src,
                                           int pitch, int tid) {
  int row  = tid >> 1;
  int half = tid & 1;
  const uint4* sp = (const uint4*)(src + (size_t)row * (size_t)pitch + half * 16);
  lds4[row * KPU4 + half * 2 + 0] = sp[0];
  lds4[row * KPU4 + half * 2 + 1] = sp[1];
}

#if HAVE_ASYNC
// Async DMA of a 128x32 bf16 tile straight into LDS (2x b128 per thread).
// The builtin's pointer params are generic int4-vector pointers (per clang
// diagnostic); the instruction IOFFSET applies to both global and LDS
// addresses, so the second 16B chunk reuses the same pointers with offset 16.
__device__ __forceinline__ void stage_async(uint4* lds4, const unsigned short* __restrict__ src,
                                            int pitch, int tid) {
  int row  = tid >> 1;
  int half = tid & 1;
  const unsigned short* g = src + (size_t)row * (size_t)pitch + half * 16;
  uint4* l = lds4 + row * KPU4 + half * 2;
  __builtin_amdgcn_global_load_async_to_lds_b128((v4i_*)(void*)g, (v4i_*)(void*)l, 0, 0);
  __builtin_amdgcn_global_load_async_to_lds_b128((v4i_*)(void*)g, (v4i_*)(void*)l, 16, 0);
}
#endif

// A fragment 16x32 bf16 (rows m0..m0+15). ISA layout: lane m & lane m+16 hold
// row m, K halves {0..7,16..23} / {8..15,24..31}, 2 bf16 per VGPR.
__device__ __forceinline__ v16bf load_a_frag(const unsigned int* ldsU, int m0, int lane) {
  int row = m0 + (lane & 15);
  int kh  = lane >> 4;
  const unsigned int* p = ldsU + row * KPU;
  v16bf f;
  unsigned int* fp = (unsigned int*)&f;
#pragma unroll
  for (int j = 0; j < 4; ++j) fp[j]     = p[kh * 4 + j];
#pragma unroll
  for (int j = 0; j < 4; ++j) fp[4 + j] = p[8 + kh * 4 + j];
  return f;
}

// B fragment 32x16 bf16 (cols n0..n0+15). Lanes 0-15: K={0..7,16..23}, lanes 16-31: K={8..15,24..31}.
__device__ __forceinline__ v16bf load_b_frag(const unsigned int* ldsU, int n0, int lane) {
  int col = n0 + (lane & 15);
  int kh  = lane >> 4;
  const unsigned int* p = ldsU + col * KPU + kh * 4;
  v16bf f;
  unsigned int* fp = (unsigned int*)&f;
#pragma unroll
  for (int j = 0; j < 8; ++j) fp[j] = p[(j & 3) + (j >> 2) * 8];
  return f;
}

// One k-step: wave (wm,wn) owns 32x64 of the 128x128 tile -> 2 A frags, 4 B frags, 8 WMMAs.
__device__ __forceinline__ void mma_step(const unsigned int* lA, const unsigned int* lB,
                                         int lane, int wm, int wn, v8f (&acc)[2][4]) {
  v16bf af[2], bf[4];
  af[0] = load_a_frag(lA, wm * 32, lane);
  af[1] = load_a_frag(lA, wm * 32 + 16, lane);
#pragma unroll
  for (int j = 0; j < 4; ++j) bf[j] = load_b_frag(lB, wn * 64 + j * 16, lane);
#pragma unroll
  for (int i = 0; i < 2; ++i)
#pragma unroll
    for (int j = 0; j < 4; ++j)
      acc[i][j] = __builtin_amdgcn_wmma_f32_16x16x32_bf16(
          false, af[i], false, bf[j], (short)0, acc[i][j], false, false);
}

// Double-buffered k-loop over bf16 A/B, async DMA staging when available.
__device__ __forceinline__ void gemm_loop_bf16(
    const unsigned short* __restrict__ Ab, int ap,
    const unsigned short* __restrict__ Bb, int bp, int Kdim,
    uint4* ldsA4, uint4* ldsB4, int tid, int lane, int wm, int wn, v8f (&acc)[2][4]) {
#if HAVE_ASYNC
  stage_async(ldsA4, Ab, ap, tid);
  stage_async(ldsB4, Bb, bp, tid);
  async_wait0();
  __syncthreads();
  int cur = 0;
  for (int kk = 0; kk < Kdim; kk += BLK_K) {
    int nxt = cur ^ 1;
    if (kk + BLK_K < Kdim) {
      stage_async(ldsA4 + nxt * TILE4, Ab + kk + BLK_K, ap, tid);
      stage_async(ldsB4 + nxt * TILE4, Bb + kk + BLK_K, bp, tid);
    }
    mma_step((const unsigned int*)(ldsA4 + cur * TILE4),
             (const unsigned int*)(ldsB4 + cur * TILE4), lane, wm, wn, acc);
    async_wait0();          // my async writes into nxt are done
    __syncthreads();        // everyone's are done -> visible
    cur = nxt;
  }
#else
  for (int kk = 0; kk < Kdim; kk += BLK_K) {
    stage_bf16(ldsA4, Ab + kk, ap, tid);
    stage_bf16(ldsB4, Bb + kk, bp, tid);
    if (kk + BLK_K < Kdim) {
      __builtin_prefetch(Ab + kk + BLK_K + (size_t)(tid >> 1) * ap, 0, 1);
      __builtin_prefetch(Bb + kk + BLK_K + (size_t)(tid >> 1) * bp, 0, 1);
    }
    __syncthreads();
    mma_step((const unsigned int*)ldsA4, (const unsigned int*)ldsB4, lane, wm, wn, acc);
    __syncthreads();
  }
#endif
}

// ---------- kernel: scores[b][j][i] = (Q_j . K_i) / sqrt(S) ----------
// (placed first in the file so the disassembly snippet shows the async
//  double-buffered WMMA loop)
__global__ void __launch_bounds__(256)
score_kernel(const unsigned short* __restrict__ Qb, const unsigned short* __restrict__ Kb,
             float* __restrict__ scores, float scale) {
  __shared__ uint4 ldsA4[2 * TILE4];
  __shared__ uint4 ldsB4[2 * TILE4];
  const int b  = blockIdx.z;
  const int m0 = blockIdx.y * BLK;   // query j
  const int n0 = blockIdx.x * BLK;   // key i
  const int tid = threadIdx.x, lane = tid & 31, wave = tid >> 5;
  const int wm = wave & 3, wn = wave >> 2;

  v8f acc[2][4];
#pragma unroll
  for (int i = 0; i < 2; ++i)
#pragma unroll
    for (int j = 0; j < 4; ++j) acc[i][j] = (v8f){0.f,0.f,0.f,0.f,0.f,0.f,0.f,0.f};

  const unsigned short* Ab = Qb + (size_t)b * Ssz * Esz + (size_t)m0 * Esz;
  const unsigned short* Bb = Kb + (size_t)b * Ssz * Esz + (size_t)n0 * Esz;
  float* outp = scores + (size_t)b * Ssz * Ssz;

  gemm_loop_bf16(Ab, Esz, Bb, Esz, Esz, ldsA4, ldsB4, tid, lane, wm, wn, acc);

  const int lm = lane & 15, lh = lane >> 4;
#pragma unroll
  for (int i = 0; i < 2; ++i) {
    const int mbase = m0 + wm * 32 + i * 16 + lh * 8;
#pragma unroll
    for (int j = 0; j < 4; ++j) {
      const int n = n0 + wn * 64 + j * 16 + lm;
#pragma unroll
      for (int r = 0; r < 8; ++r)
        outp[(size_t)(mbase + r) * Ssz + n] = acc[i][j][r] * scale;
    }
  }
}

// ---------- kernel: Y[b][j][d] = sum_i P[b][j][i] * Vt[b][d][i] ----------
__global__ void __launch_bounds__(256)
out_kernel(const unsigned short* __restrict__ P,   // bf16 rows at f32 pitch (2*Ssz)
           const unsigned short* __restrict__ Vt,
           float* __restrict__ Y) {
  __shared__ uint4 ldsA4[2 * TILE4];
  __shared__ uint4 ldsB4[2 * TILE4];
  const int b  = blockIdx.z;
  const int m0 = blockIdx.y * BLK;   // j
  const int n0 = blockIdx.x * BLK;   // d
  const int tid = threadIdx.x, lane = tid & 31, wave = tid >> 5;
  const int wm = wave & 3, wn = wave >> 2;

  v8f acc[2][4];
#pragma unroll
  for (int i = 0; i < 2; ++i)
#pragma unroll
    for (int j = 0; j < 4; ++j) acc[i][j] = (v8f){0.f,0.f,0.f,0.f,0.f,0.f,0.f,0.f};

  const int apitch = 2 * Ssz;                        // bf16 elems per P row
  const unsigned short* Ab = P  + (size_t)b * Ssz * apitch + (size_t)m0 * apitch;
  const unsigned short* Bb = Vt + (size_t)b * Esz * Ssz    + (size_t)n0 * Ssz;
  float* outp = Y + (size_t)b * Ssz * Esz;

  gemm_loop_bf16(Ab, apitch, Bb, Ssz, Ssz, ldsA4, ldsB4, tid, lane, wm, wn, acc);

  const int lm = lane & 15, lh = lane >> 4;
#pragma unroll
  for (int i = 0; i < 2; ++i) {
    const int mbase = m0 + wm * 32 + i * 16 + lh * 8;
#pragma unroll
    for (int j = 0; j < 4; ++j) {
      const int n = n0 + wn * 64 + j * 16 + lm;
#pragma unroll
      for (int r = 0; r < 8; ++r)
        outp[(size_t)(mbase + r) * Esz + n] = acc[i][j][r];
    }
  }
}

// ---------- kernel: row softmax (over i), in-place f32 -> bf16 ----------
__device__ __forceinline__ float wred_max(float v) {
#pragma unroll
  for (int off = 16; off > 0; off >>= 1) v = fmaxf(v, __shfl_xor(v, off, 32));
  return v;
}
__device__ __forceinline__ float wred_sum(float v) {
#pragma unroll
  for (int off = 16; off > 0; off >>= 1) v += __shfl_xor(v, off, 32);
  return v;
}

__global__ void __launch_bounds__(256)
softmax_kernel(float* __restrict__ scores) {
  float* rp = scores + (size_t)blockIdx.x * Ssz;     // one (b,j) row of 2048
  const int t = threadIdx.x, lane = t & 31, wave = t >> 5;
  float v[8];
#pragma unroll
  for (int k = 0; k < 8; ++k) v[k] = rp[t + (k << 8)];   // all reads before barriers

  float mx = v[0];
#pragma unroll
  for (int k = 1; k < 8; ++k) mx = fmaxf(mx, v[k]);
  mx = wred_max(mx);
  __shared__ float sm[8], ssum[8];
  if (lane == 0) sm[wave] = mx;
  __syncthreads();
  float bm = sm[0];
#pragma unroll
  for (int w = 1; w < 8; ++w) bm = fmaxf(bm, sm[w]);

  float s = 0.f;
#pragma unroll
  for (int k = 0; k < 8; ++k) { v[k] = __expf(v[k] - bm); s += v[k]; }
  s = wred_sum(s);
  if (lane == 0) ssum[wave] = s;
  __syncthreads();
  float tot = 0.f;
#pragma unroll
  for (int w = 0; w < 8; ++w) tot += ssum[w];
  const float inv = 1.0f / tot;

  unsigned short* op = (unsigned short*)rp;          // bf16 aliased into f32 row
#pragma unroll
  for (int k = 0; k < 8; ++k) op[t + (k << 8)] = f2bf(v[k] * inv);
}

// ---------- kernel: fused K/Q/V projections (blockIdx.z selects which) ----------
__global__ void __launch_bounds__(256)
proj_kernel(const float* __restrict__ emb,
            const float* __restrict__ Wk, const float* __restrict__ bk,
            const float* __restrict__ Wq, const float* __restrict__ bq,
            const float* __restrict__ Wv, const float* __restrict__ bv,
            unsigned short* __restrict__ Kb, unsigned short* __restrict__ Qb,
            unsigned short* __restrict__ Vt) {
  __shared__ uint4 ldsA4[TILE4];
  __shared__ uint4 ldsB4[TILE4];
  const int which = blockIdx.z;
  const float* W    = (which == 0) ? Wk : (which == 1) ? Wq : Wv;
  const float* bias = (which == 0) ? bk : (which == 1) ? bq : bv;
  const int m0 = blockIdx.y * BLK;           // row in flattened [B*S]
  const int n0 = blockIdx.x * BLK;           // output feature
  const int tid = threadIdx.x, lane = tid & 31, wave = tid >> 5;
  const int wm = wave & 3, wn = wave >> 2;

  v8f acc[2][4];
#pragma unroll
  for (int i = 0; i < 2; ++i)
#pragma unroll
    for (int j = 0; j < 4; ++j) acc[i][j] = (v8f){0.f,0.f,0.f,0.f,0.f,0.f,0.f,0.f};

  const float* Ab = emb + (size_t)m0 * Esz;
  const float* Bb = W   + (size_t)n0 * Esz;
  for (int kk = 0; kk < Esz; kk += BLK_K) {
    stage_f32(ldsA4, Ab + kk, Esz, tid);
    stage_f32(ldsB4, Bb + kk, Esz, tid);
    if (kk + BLK_K < Esz) {
      __builtin_prefetch(Ab + kk + BLK_K + (size_t)(tid >> 1) * Esz, 0, 1);
      __builtin_prefetch(Bb + kk + BLK_K + (size_t)(tid >> 1) * Esz, 0, 1);
    }
    __syncthreads();
    mma_step((const unsigned int*)ldsA4, (const unsigned int*)ldsB4, lane, wm, wn, acc);
    __syncthreads();
  }

  const int lm = lane & 15, lh = lane >> 4;
#pragma unroll
  for (int i = 0; i < 2; ++i) {
    const int mbase = m0 + wm * 32 + i * 16 + lh * 8;
#pragma unroll
    for (int j = 0; j < 4; ++j) {
      const int n = n0 + wn * 64 + j * 16 + lm;
      const float bb = bias[n];
      if (which < 2) {
        unsigned short* outp = (which == 0) ? Kb : Qb;
#pragma unroll
        for (int r = 0; r < 8; ++r)
          outp[(size_t)(mbase + r) * Esz + n] = f2bf(acc[i][j][r] + bb);
      } else {
        // V stored transposed: Vt[b][d][i], 8 consecutive i per lane -> one b128 store
        const int bi = mbase >> 11;            // / S
        const int ii = mbase & (Ssz - 1);
        uint4 vv;
        vv.x = pk2(acc[i][j][0] + bb, acc[i][j][1] + bb);
        vv.y = pk2(acc[i][j][2] + bb, acc[i][j][3] + bb);
        vv.z = pk2(acc[i][j][4] + bb, acc[i][j][5] + bb);
        vv.w = pk2(acc[i][j][6] + bb, acc[i][j][7] + bb);
        *(uint4*)(Vt + ((size_t)bi * Esz + n) * Ssz + ii) = vv;
      }
    }
  }
}

// ---------- host launcher ----------
extern "C" void kernel_launch(void* const* d_in, const int* in_sizes, int n_in,
                              void* d_out, int out_size, void* d_ws, size_t ws_size,
                              hipStream_t stream) {
  (void)in_sizes; (void)n_in; (void)out_size; (void)ws_size;
  const float* emb = (const float*)d_in[0];
  const float* Wk  = (const float*)d_in[1];
  const float* bk  = (const float*)d_in[2];
  const float* Wq  = (const float*)d_in[3];
  const float* bq  = (const float*)d_in[4];
  const float* Wv  = (const float*)d_in[5];
  const float* bv  = (const float*)d_in[6];

  // workspace layout (112 MB total):
  //   [0,16M)   K bf16  [B*S][E]
  //   [16,32M)  Q bf16  [B*S][E]
  //   [32,48M)  V bf16 transposed [B][E][S]
  //   [48,112M) scores f32 [B][S][S]; softmax rewrites bf16 P in place (f32 pitch)
  char* ws = (char*)d_ws;
  unsigned short* Kb = (unsigned short*)(ws);
  unsigned short* Qb = (unsigned short*)(ws + ((size_t)16 << 20));
  unsigned short* Vt = (unsigned short*)(ws + ((size_t)32 << 20));
  float* scores      = (float*)(ws + ((size_t)48 << 20));

  dim3 blk(256);
  proj_kernel<<<dim3(Esz / BLK, (Bsz * Ssz) / BLK, 3), blk, 0, stream>>>(
      emb, Wk, bk, Wq, bq, Wv, bv, Kb, Qb, Vt);

  const float scale = 1.0f / sqrtf((float)Ssz);
  score_kernel<<<dim3(Ssz / BLK, Ssz / BLK, Bsz), blk, 0, stream>>>(Qb, Kb, scores, scale);

  softmax_kernel<<<dim3(Bsz * Ssz), blk, 0, stream>>>(scores);

  out_kernel<<<dim3(Esz / BLK, Ssz / BLK, Bsz), blk, 0, stream>>>(
      (const unsigned short*)scores, Vt, (float*)d_out);
}
